// LayerNormLSTM_15607911153962
// MI455X (gfx1250) — matrix operand order repaired and checked by reading, not compiled, not run on hardware
//
#include <hip/hip_runtime.h>
#include <hip/hip_bf16.h>
#include <math.h>

typedef __attribute__((ext_vector_type(16))) _Float16 v16h;
typedef __attribute__((ext_vector_type(8)))  _Float16 v8h;
typedef __attribute__((ext_vector_type(8)))  float    v8f;

#define T_STEPS 256
#define BATCH   64
#define HDIM    512
#define G4      2048   // 4*H gate width
#define LAYERS  2
#define HS      544    // padded LDS row stride for h (halfs)

__device__ __forceinline__ float sigm(float x) { return 1.0f / (1.0f + __expf(-x)); }

__device__ __forceinline__ v8f wmma_f16(v16h a, v16h b, v8f c) {
  // D = A(16x32 f16) * B(32x16 f16) + C(16x16 f32)
  return __builtin_amdgcn_wmma_f32_16x16x32_f16(false, a, false, b, (short)0, c, false, false);
}

__device__ __forceinline__ v16h cat16(v8h lo, v8h hi) {
  return __builtin_shufflevector(lo, hi, 0,1,2,3,4,5,6,7,8,9,10,11,12,13,14,15);
}

// ---------------- f32 -> f16 conversion ----------------
__global__ void k_cvt_f16(const float* __restrict__ src, _Float16* __restrict__ dst, int n) {
  int i = blockIdx.x * blockDim.x + threadIdx.x;
  if (i < n) dst[i] = (_Float16)src[i];
}

// ---------------- Fused input-projection GEMM + row LayerNorm ----------------
// out[row, 0..2048) = LN(A16[row,:] @ W16^T + bias) * g + be
// One block (1024 thr = 32 waves) owns a full 16-row x 2048-col stripe:
// wave w computes cols [w*64, w*64+64) -> LN stats reduced from registers.
__global__ void __launch_bounds__(1024)
k_xproj_ln(const _Float16* __restrict__ A16,
           const _Float16* __restrict__ W16,
           const float* __restrict__ bias,
           const float* __restrict__ g,
           const float* __restrict__ be,
           float* __restrict__ out) {
  __shared__ float redS[32 * 16];
  __shared__ float redQ[32 * 16];
  __shared__ float s_mean[16];
  __shared__ float s_rstd[16];

  const int tid  = threadIdx.x;
  const int wave = tid >> 5;
  const int lane = tid & 31;
  const int half = lane >> 4;
  const int l15  = lane & 15;
  const int rowbase = blockIdx.x * 16;
  const int wbase   = wave * 64;

  const _Float16* arow = A16 + (size_t)(rowbase + l15) * HDIM + half * 16;
  const _Float16* wrow = W16 + (size_t)(wbase + l15) * HDIM + half * 16;

  v8f acc0 = {}, acc1 = {}, acc2 = {}, acc3 = {};
  // software-pipelined K loop (double-buffered fragments)
  v16h a_cur = cat16(*(const v8h*)(arow), *(const v8h*)(arow + 8));
  v16h b0 = *(const v16h*)(wrow);
  v16h b1 = *(const v16h*)(wrow + 16 * HDIM);
  v16h b2 = *(const v16h*)(wrow + 32 * HDIM);
  v16h b3 = *(const v16h*)(wrow + 48 * HDIM);
  #pragma unroll
  for (int kk = 0; kk < 16; ++kk) {
    v16h a_n, c0, c1, c2, c3;
    if (kk < 15) {
      const int kn = (kk + 1) * 32;
      a_n = cat16(*(const v8h*)(arow + kn), *(const v8h*)(arow + kn + 8));
      c0 = *(const v16h*)(wrow + kn);
      c1 = *(const v16h*)(wrow + 16 * HDIM + kn);
      c2 = *(const v16h*)(wrow + 32 * HDIM + kn);
      c3 = *(const v16h*)(wrow + 48 * HDIM + kn);
    }
    acc0 = wmma_f16(a_cur, b0, acc0);
    acc1 = wmma_f16(a_cur, b1, acc1);
    acc2 = wmma_f16(a_cur, b2, acc2);
    acc3 = wmma_f16(a_cur, b3, acc3);
    if (kk < 15) { a_cur = a_n; b0 = c0; b1 = c1; b2 = c2; b3 = c3; }
  }

  // add bias (pre-LN, matches reference)
  const float bv0 = bias[wbase + l15];
  const float bv1 = bias[wbase + 16 + l15];
  const float bv2 = bias[wbase + 32 + l15];
  const float bv3 = bias[wbase + 48 + l15];
  #pragma unroll
  for (int v = 0; v < 8; ++v) {
    acc0[v] += bv0; acc1[v] += bv1; acc2[v] += bv2; acc3[v] += bv3;
  }

  // per-row partial sums from registers: lanes 0-15 hold rows 0-7 (VGPR v),
  // lanes 16-31 hold rows 8-15; shfl_xor(1..8) reduces within each 16-lane half.
  #pragma unroll
  for (int v = 0; v < 8; ++v) {
    float s = acc0[v] + acc1[v] + acc2[v] + acc3[v];
    float q = acc0[v]*acc0[v] + acc1[v]*acc1[v] + acc2[v]*acc2[v] + acc3[v]*acc3[v];
    #pragma unroll
    for (int off = 8; off >= 1; off >>= 1) {
      s += __shfl_xor(s, off, 32); q += __shfl_xor(q, off, 32);
    }
    if (l15 == 0) {
      redS[wave * 16 + half * 8 + v] = s;
      redQ[wave * 16 + half * 8 + v] = q;
    }
  }
  __syncthreads();
  if (tid < 16) {
    float s = 0.f, q = 0.f;
    for (int w = 0; w < 32; ++w) { s += redS[w * 16 + tid]; q += redQ[w * 16 + tid]; }
    float m = s * (1.0f / G4);
    s_mean[tid] = m;
    s_rstd[tid] = rsqrtf(q * (1.0f / G4) - m * m + 1e-5f);
  }
  __syncthreads();

  // normalize, scale, write
  const float g0 = g[wbase + l15],      e0 = be[wbase + l15];
  const float g1 = g[wbase + 16 + l15], e1 = be[wbase + 16 + l15];
  const float g2 = g[wbase + 32 + l15], e2 = be[wbase + 32 + l15];
  const float g3 = g[wbase + 48 + l15], e3 = be[wbase + 48 + l15];
  #pragma unroll
  for (int v = 0; v < 8; ++v) {
    const int m = v + half * 8;
    const float mn = s_mean[m], rs = s_rstd[m];
    float* orow = out + (size_t)(rowbase + m) * G4 + wbase + l15;
    orow[0]  = (acc0[v] - mn) * rs * g0 + e0;
    orow[16] = (acc1[v] - mn) * rs * g1 + e1;
    orow[32] = (acc2[v] - mn) * rs * g2 + e2;
    orow[48] = (acc3[v] - mn) * rs * g3 + e3;
  }
}

// ---------------- Persistent LSTM recurrence ----------------
// grid = 4 (batch chunks of 16), block = 1024 (32 waves).
// xg already contains LN(x@Wih^T + b_ih)*g_ih + be_ih.
__global__ void __launch_bounds__(1024)
k_lstm_seq(const float* __restrict__ xg,
           const _Float16* __restrict__ Whh,  // [2048,512] f16
           const float* __restrict__ bhh,
           const float* __restrict__ ghh,
           const float* __restrict__ behh,
           const float* __restrict__ gho,
           const float* __restrict__ beho,
           float* __restrict__ y,    // [T*B, 512]
           float* __restrict__ hT,   // [B=64 rows of this layer's final h]
           float* __restrict__ cT) {
  extern __shared__ char smem_raw[];
  float*    gates = (float*)smem_raw;               // 16*2048 f32
  float*    cbuf  = gates + 16 * G4;                // 16*512  f32
  _Float16* hbuf  = (_Float16*)(cbuf + 16 * HDIM);  // 16*HS   f16
  float*    mean  = (float*)(hbuf + 16 * HS);       // 16
  float*    rstd  = mean + 16;                      // 16

  const int tid  = threadIdx.x;
  const int wave = tid >> 5;
  const int lane = tid & 31;
  const int half = lane >> 4;
  const int l15  = lane & 15;
  const int bc   = blockIdx.x;   // batch chunk: rows bc*16 .. bc*16+15
  const int nbase = wave * 64;   // 64 gate columns per wave

  const _Float16* wrow = Whh + (size_t)(nbase + l15) * HDIM + half * 16;
  const _Float16* hrow = hbuf + l15 * HS + half * 16;

  for (int i = tid; i < 16 * HDIM; i += 1024) cbuf[i] = 0.f;
  for (int i = tid; i < 16 * HS;   i += 1024) hbuf[i] = (_Float16)0.f;
  __syncthreads();

  for (int t = 0; t < T_STEPS; ++t) {
    // ---- Phase A: gates_pre = h @ Whh^T + bhh (WMMA, pipelined) ----
    v8f acc0 = {}, acc1 = {}, acc2 = {}, acc3 = {};
    v16h a_cur = cat16(*(const v8h*)(hrow), *(const v8h*)(hrow + 8));
    v16h b0 = *(const v16h*)(wrow);
    v16h b1 = *(const v16h*)(wrow + 16 * HDIM);
    v16h b2 = *(const v16h*)(wrow + 32 * HDIM);
    v16h b3 = *(const v16h*)(wrow + 48 * HDIM);
    #pragma unroll
    for (int kk = 0; kk < 16; ++kk) {
      v16h a_n, c0, c1, c2, c3;
      if (kk < 15) {
        const int kn = (kk + 1) * 32;
        a_n = cat16(*(const v8h*)(hrow + kn), *(const v8h*)(hrow + kn + 8));
        c0 = *(const v16h*)(wrow + kn);
        c1 = *(const v16h*)(wrow + 16 * HDIM + kn);
        c2 = *(const v16h*)(wrow + 32 * HDIM + kn);
        c3 = *(const v16h*)(wrow + 48 * HDIM + kn);
      }
      acc0 = wmma_f16(a_cur, b0, acc0);
      acc1 = wmma_f16(a_cur, b1, acc1);
      acc2 = wmma_f16(a_cur, b2, acc2);
      acc3 = wmma_f16(a_cur, b3, acc3);
      if (kk < 15) { a_cur = a_n; b0 = c0; b1 = c1; b2 = c2; b3 = c3; }
    }
    {
      v8f accs[4] = {acc0, acc1, acc2, acc3};
      #pragma unroll
      for (int j = 0; j < 4; ++j) {
        int n = nbase + j * 16 + l15;
        float bv = bhh[n];
        #pragma unroll
        for (int v = 0; v < 8; ++v)
          gates[(v + half * 8) * G4 + n] = accs[j][v] + bv;
      }
    }
    __syncthreads();

    // ---- Phase B: LN stats over 2048 gate cols per row ----
    if (wave < 16) {
      float s = 0.f, q = 0.f;
      for (int i = 0; i < 64; ++i) {
        float v = gates[wave * G4 + lane + i * 32]; s += v; q += v * v;
      }
      #pragma unroll
      for (int off = 16; off >= 1; off >>= 1) {
        s += __shfl_xor(s, off, 32); q += __shfl_xor(q, off, 32);
      }
      if (lane == 0) {
        float m = s * (1.0f / G4);
        mean[wave] = m;
        rstd[wave] = rsqrtf(q * (1.0f / G4) - m * m + 1e-5f);
      }
    }
    __syncthreads();

    // ---- Phase C: LN-scale, add precomputed x-gates, activate ----
    const float* xgt = xg + ((size_t)t * BATCH + bc * 16) * G4;
    #pragma unroll
    for (int p = 0; p < 32; ++p) {
      int idx = tid + p * 1024;
      int m = idx >> 11, n = idx & (G4 - 1);
      float val = (gates[idx] - mean[m]) * rstd[m] * ghh[n] + behh[n] + xgt[idx];
      gates[idx] = (n < 3 * HDIM) ? sigm(val) : tanhf(val);
    }
    __syncthreads();

    // ---- Phase D: c update ----
    const bool last = (t == T_STEPS - 1);
    #pragma unroll
    for (int p = 0; p < 8; ++p) {
      int idx = tid + p * 1024;
      int m = idx >> 9, j = idx & (HDIM - 1);
      float iv = gates[m * G4 + j];
      float fv = gates[m * G4 + HDIM + j];
      float gv = gates[m * G4 + 3 * HDIM + j];
      float cv = fv * cbuf[idx] + iv * gv;
      cbuf[idx] = cv;
      if (last) cT[(size_t)(bc * 16 + m) * HDIM + j] = cv;
    }
    __syncthreads();

    // ---- Phase E: LN stats over c (512 per row) ----
    if (wave < 16) {
      float s = 0.f, q = 0.f;
      for (int i = 0; i < 16; ++i) {
        float v = cbuf[wave * HDIM + lane + i * 32]; s += v; q += v * v;
      }
      #pragma unroll
      for (int off = 16; off >= 1; off >>= 1) {
        s += __shfl_xor(s, off, 32); q += __shfl_xor(q, off, 32);
      }
      if (lane == 0) {
        float m = s * (1.0f / HDIM);
        mean[wave] = m;
        rstd[wave] = rsqrtf(q * (1.0f / HDIM) - m * m + 1e-5f);
      }
    }
    __syncthreads();

    // ---- Phase F: h = o * tanh(LN(c)) ----
    float* yt = y + ((size_t)t * BATCH + bc * 16) * HDIM;
    #pragma unroll
    for (int p = 0; p < 8; ++p) {
      int idx = tid + p * 1024;
      int m = idx >> 9, j = idx & (HDIM - 1);
      float ov = gates[m * G4 + 2 * HDIM + j];
      float cn = (cbuf[idx] - mean[m]) * rstd[m] * gho[j] + beho[j];
      float hv = ov * tanhf(cn);
      hbuf[m * HS + j] = (_Float16)hv;
      yt[idx] = hv;
      if (last) hT[(size_t)(bc * 16 + m) * HDIM + j] = hv;
    }
    __syncthreads();
  }
}

#define SMEM_BYTES (16 * G4 * 4 + 16 * HDIM * 4 + 16 * HS * 2 + 64 * 4)

extern "C" void kernel_launch(void* const* d_in, const int* in_sizes, int n_in,
                              void* d_out, int out_size, void* d_ws, size_t ws_size,
                              hipStream_t stream) {
  (void)in_sizes; (void)n_in; (void)out_size; (void)ws_size;
  const float* x = (const float*)d_in[0];

  float* out      = (float*)d_out;
  float* y_final  = out;                                          // [T,B,H]
  float* hT_all   = out + (size_t)T_STEPS * BATCH * HDIM;         // [L,B,H]
  float* cT_all   = hT_all + (size_t)LAYERS * BATCH * HDIM;       // [L,B,H]

  char* ws = (char*)d_ws;
  _Float16* Wih16 = (_Float16*)ws;  ws += (size_t)G4 * HDIM * 2;
  _Float16* Whh16 = (_Float16*)ws;  ws += (size_t)G4 * HDIM * 2;
  _Float16* act16 = (_Float16*)ws;  ws += (size_t)T_STEPS * BATCH * HDIM * 2;
  float*    xgbuf = (float*)ws;     ws += (size_t)T_STEPS * BATCH * G4 * 4;
  float*    y1    = (float*)ws;     ws += (size_t)T_STEPS * BATCH * HDIM * 4;

  (void)hipFuncSetAttribute(reinterpret_cast<const void*>(k_lstm_seq),
                            hipFuncAttributeMaxDynamicSharedMemorySize, SMEM_BYTES);

  const int MROWS = T_STEPS * BATCH;           // 16384
  const int NW    = G4 * HDIM;                 // weight elems
  const int NA    = MROWS * HDIM;              // activation elems

  for (int l = 0; l < LAYERS; ++l) {
    const float* W_ih = (const float*)d_in[1 + 10 * l + 0];
    const float* W_hh = (const float*)d_in[1 + 10 * l + 1];
    const float* b_ih = (const float*)d_in[1 + 10 * l + 2];
    const float* b_hh = (const float*)d_in[1 + 10 * l + 3];
    const float* g_ih = (const float*)d_in[1 + 10 * l + 4];
    const float* be_ih= (const float*)d_in[1 + 10 * l + 5];
    const float* g_hh = (const float*)d_in[1 + 10 * l + 6];
    const float* be_hh= (const float*)d_in[1 + 10 * l + 7];
    const float* g_ho = (const float*)d_in[1 + 10 * l + 8];
    const float* be_ho= (const float*)d_in[1 + 10 * l + 9];

    const float* src = (l == 0) ? x : y1;

    k_cvt_f16<<<(NW + 255) / 256, 256, 0, stream>>>(W_ih, Wih16, NW);
    k_cvt_f16<<<(NW + 255) / 256, 256, 0, stream>>>(W_hh, Whh16, NW);
    k_cvt_f16<<<(NA + 255) / 256, 256, 0, stream>>>(src, act16, NA);

    // xgbuf = LN(act @ Wih^T + b_ih)*g_ih + be_ih, fused (one stripe per block)
    k_xproj_ln<<<MROWS / 16, 1024, 0, stream>>>(act16, Wih16, b_ih, g_ih, be_ih, xgbuf);

    float* yout = (l == 0) ? y1 : y_final;
    k_lstm_seq<<<4, 1024, SMEM_BYTES, stream>>>(
        xgbuf, Whh16, b_hh, g_hh, be_hh, g_ho, be_ho,
        yout, hT_all + (size_t)l * BATCH * HDIM, cT_all + (size_t)l * BATCH * HDIM);
  }
}
